// ImuNet_35914516529193
// MI455X (gfx1250) — matrix operand order repaired and checked by reading, not compiled, not run on hardware
//
#include <hip/hip_runtime.h>
#include <hip/hip_bf16.h>

typedef __attribute__((ext_vector_type(16))) _Float16 v16h;
typedef __attribute__((ext_vector_type(8)))  float    v8f;

#define LEAK   0.1f
#define BN_EPS 1e-3f
#define BB   128      // batch
#define TT   1024     // time steps
#define HID  256      // LSTM hidden

__device__ __forceinline__ float sigmoidf_(float x) { return 1.0f / (1.0f + __expf(-x)); }

// ===========================================================================
// Packed tile layouts (fragments contiguous per lane -> b128 ld/st):
//  A-tile (16Mx32K f16): tile(mt,kt) = 32 lanes x 16 halves.
//  B-tile (32Kx16N f16): tile(kt,nt) = 32 lanes x 16 halves.
//  C-tile (16Mx16N f32): tile(mt,nt) = 32 lanes x 8 floats.
// ===========================================================================

// ---- prep: f32->f16 weights packed in B-tile layout; fold BN -------------
__global__ void k_prep(const float* W1, const float* U1, const float* W2, const float* U2,
                       const float* g1, const float* b1, const float* m1, const float* v1,
                       const float* g2, const float* b2, const float* m2, const float* v2,
                       const float* g3, const float* b3, const float* m3, const float* v3,
                       _Float16* W1p, _Float16* U1p, _Float16* W2p, _Float16* U2p,
                       float* s1, float* t1, float* s2, float* t2, float* s3, float* t3) {
    int j = blockIdx.x * blockDim.x + threadIdx.x;
    const int NW = HID * 4 * HID;  // 262144 halves per matrix
    if (j < NW) {
        int e    = j & 15;
        int lane = (j >> 4) & 31;
        int tidx = j >> 9;          // 512 halves per tile
        int nt   = tidx & 63;       // 64 N-tiles (N=1024)
        int kt   = tidx >> 6;       // 8 K-tiles  (K=256)
        int krow = kt * 32 + ((e >> 3) << 4) + ((lane >> 4) << 3) + (e & 7);
        int col  = nt * 16 + (lane & 15);
        int src  = krow * 1024 + col;
        W1p[j] = (_Float16)W1[src];
        U1p[j] = (_Float16)U1[src];
        W2p[j] = (_Float16)W2[src];
        U2p[j] = (_Float16)U2[src];
    }
    if (j < 64)  { float s = g1[j] * rsqrtf(v1[j] + BN_EPS); s1[j] = s; t1[j] = b1[j] - m1[j] * s; }
    if (j < 128) { float s = g2[j] * rsqrtf(v2[j] + BN_EPS); s2[j] = s; t2[j] = b2[j] - m2[j] * s; }
    if (j < 256) { float s = g3[j] * rsqrtf(v3[j] + BN_EPS); s3[j] = s; t3[j] = b3[j] - m3[j] * s; }
}

// ---- conv encoder (direct, SAME pad, fused BN+leaky) ---------------------
__global__ void k_conv1(const float* __restrict__ x, const float* __restrict__ k,
                        const float* __restrict__ s, const float* __restrict__ tb,
                        float* __restrict__ y) {
    int idx = blockIdx.x * blockDim.x + threadIdx.x;  // B*T*64
    int co = idx & 63; int rem = idx >> 6; int t = rem & (TT - 1); int b = rem >> 10;
    float acc = 0.f;
#pragma unroll
    for (int dt = 0; dt < 3; ++dt) {
        int ts = t + dt - 1;
        if (ts >= 0 && ts < TT) {
#pragma unroll
            for (int ci = 0; ci < 6; ++ci)
                acc = fmaf(x[((size_t)b * TT + ts) * 6 + ci], k[(dt * 6 + ci) * 64 + co], acc);
        }
    }
    float ya = fmaf(acc, s[co], tb[co]);
    y[idx] = ya > 0.f ? ya : LEAK * ya;
}

__global__ void k_conv2(const float* __restrict__ x, const float* __restrict__ k,
                        const float* __restrict__ s, const float* __restrict__ tb,
                        float* __restrict__ y) {
    int idx = blockIdx.x * blockDim.x + threadIdx.x;  // B*T*128
    int co = idx & 127; int rem = idx >> 7; int t = rem & (TT - 1); int b = rem >> 10;
    float acc = 0.f;
    for (int dt = 0; dt < 3; ++dt) {
        int ts = t + dt - 1;
        if (ts >= 0 && ts < TT) {
            const float* xr = x + ((size_t)b * TT + ts) * 64;
#pragma unroll 8
            for (int ci = 0; ci < 64; ++ci)
                acc = fmaf(xr[ci], k[(dt * 64 + ci) * 128 + co], acc);
        }
    }
    float ya = fmaf(acc, s[co], tb[co]);
    y[idx] = ya > 0.f ? ya : LEAK * ya;
}

// conv3 writes f16 directly in packed A-tile layout (rows = t*B + b, cols = co)
__global__ void k_conv3(const float* __restrict__ x, const float* __restrict__ k,
                        const float* __restrict__ s, const float* __restrict__ tb,
                        _Float16* __restrict__ y) {
    int idx = blockIdx.x * blockDim.x + threadIdx.x;  // B*T*256
    int co = idx & 255; int rem = idx >> 8; int t = rem & (TT - 1); int b = rem >> 10;
    float acc = 0.f;
    for (int dt = 0; dt < 3; ++dt) {
        int ts = t + dt - 1;
        if (ts >= 0 && ts < TT) {
            const float* xr = x + ((size_t)b * TT + ts) * 128;
#pragma unroll 8
            for (int ci = 0; ci < 128; ++ci)
                acc = fmaf(xr[ci], k[(dt * 128 + ci) * 256 + co], acc);
        }
    }
    float ya = fmaf(acc, s[co], tb[co]);
    ya = ya > 0.f ? ya : LEAK * ya;
    int row = t * BB + b;
    int mt = row >> 4, r = row & 15;
    int kt = co >> 5, kl = co & 31;
    int lane = ((kl >> 3) & 1) * 16 + r;
    int v    = ((kl >> 4) << 2) | ((kl >> 1) & 3);
    int e    = (v << 1) | (kl & 1);
    y[(((size_t)mt * 8 + kt) * 32 + lane) * 16 + e] = (_Float16)ya;
}

// ---- Z = Xp @ W1p + bl1 -> f32, stored C-tile packed ---------------------
// Software-pipelined: k-step (kt+1)'s A/B fragments load during kt's WMMAs.
__global__ __launch_bounds__(256) void k_xw1(const _Float16* __restrict__ Xp,
                                             const _Float16* __restrict__ W1p,
                                             const float* __restrict__ bl1,
                                             float* __restrict__ Zp) {
    int lane = threadIdx.x & 31;
    int wave = threadIdx.x >> 5;
    int job  = blockIdx.x * 8 + wave;   // 8192 M-tiles * 16 N-groups
    int mt = job >> 4, ng = job & 15;
    int nl = lane & 15;

    const _Float16* Abase = Xp  + ((size_t)mt * 8 * 32 + lane) * 16;          // + kt*512
    const _Float16* Bbase = W1p + (((size_t)ng * 4) * 32 + lane) * 16;        // + kt*32768 + j*512

    v8f acc[4];
#pragma unroll
    for (int j = 0; j < 4; ++j) acc[j] = v8f{};

    v16h a_cur = *(const v16h*)(Abase);
    v16h b_cur[4];
#pragma unroll
    for (int j = 0; j < 4; ++j) b_cur[j] = *(const v16h*)(Bbase + (size_t)j * 512);

#pragma unroll
    for (int kt = 0; kt < 8; ++kt) {
        v16h a_nxt = a_cur;
        v16h b_nxt[4];
#pragma unroll
        for (int j = 0; j < 4; ++j) b_nxt[j] = b_cur[j];
        if (kt < 7) {
            a_nxt = *(const v16h*)(Abase + (size_t)(kt + 1) * 512);
#pragma unroll
            for (int j = 0; j < 4; ++j)
                b_nxt[j] = *(const v16h*)(Bbase + (size_t)(kt + 1) * 32768 + (size_t)j * 512);
        }
#pragma unroll
        for (int j = 0; j < 4; ++j)
            acc[j] = __builtin_amdgcn_wmma_f32_16x16x32_f16(false, a_cur, false, b_cur[j],
                                                            (short)0, acc[j], false, false);
        a_cur = a_nxt;
#pragma unroll
        for (int j = 0; j < 4; ++j) b_cur[j] = b_nxt[j];
    }
#pragma unroll
    for (int j = 0; j < 4; ++j) {
        float bias = bl1[(ng * 4 + j) * 16 + nl];
        v8f o = acc[j];
#pragma unroll
        for (int r = 0; r < 8; ++r) o[r] += bias;
        *(v8f*)(Zp + (((size_t)mt * 64 + ng * 4 + j) * 32 + lane) * 8) = o;
    }
}

// ---- persistent sequential 2-layer LSTM on a 16-WG cluster ---------------
// block = hidden block hb (0..15); wave = batch tile mt (0..7).
// 96 KB of this block's B-tiles (U1/W2/U2 columns) staged in LDS once.
__global__ __launch_bounds__(256) void k_lstm(const float* __restrict__ Zp,
                                              const _Float16* __restrict__ U1p,
                                              const _Float16* __restrict__ W2p,
                                              const _Float16* __restrict__ U2p,
                                              const float* __restrict__ bl2,
                                              _Float16* h1a, _Float16* h1b,
                                              _Float16* h2a, _Float16* h2b,
                                              float* __restrict__ out) {
    __shared__ _Float16 smem[96 * 512];   // 96 tiles x 1KB = 96 KB LDS

    int lane = threadIdx.x & 31;
    int mt   = threadIdx.x >> 5;          // batch tile 0..7
    int hb   = blockIdx.x;                // hidden block 0..15
    int m0   = mt * 16;
    int nl   = lane & 15, mo = (lane >> 4) * 8;
    int col16 = hb * 16 + nl;

    // ---- stage this block's 96 B-tiles into LDS: [phase 0:U1 1:W2 2:U2][kt][g]
    for (int i = threadIdx.x; i < 96 * 64; i += 256) {   // uint4 = 8 halves
        int tile = i >> 6, sub = i & 63;
        int phase = tile >> 5;
        int local = tile & 31;
        int kt = local >> 2, g = local & 3;
        const _Float16* base = (phase == 0) ? U1p : ((phase == 1) ? W2p : U2p);
        const uint4* src = (const uint4*)(base + (size_t)(kt * 64 + g * 16 + hb) * 512);
        ((uint4*)smem)[i] = src[sub];
    }
    __syncthreads();

    v8f c1 = {}, c2 = {}, ysum = {};
    float b2g[4];
#pragma unroll
    for (int g = 0; g < 4; ++g) b2g[g] = bl2[g * HID + col16];

    // packed-A scatter indices for h stores (derived from A-tile layout)
    int kt_h  = hb >> 1;
    int laneS = ((nl >> 3) & 1) * 16;
    int vS    = ((hb & 1) << 2) | ((nl >> 1) & 3);
    int eS    = (vS << 1) | (nl & 1);

    _Float16* h1buf[2] = {h1a, h1b};
    _Float16* h2buf[2] = {h2a, h2b};
    const size_t laneA = ((size_t)mt * 8 * 32 + lane) * 16;   // + kt*512
    const _Float16* smemL = smem + lane * 16;                 // + tile*512

    for (int t = 0; t < TT; ++t) {
        _Float16* h1_prev = h1buf[t & 1];
        _Float16* h1_cur  = h1buf[(t & 1) ^ 1];
        _Float16* h2_prev = h2buf[t & 1];
        _Float16* h2_cur  = h2buf[(t & 1) ^ 1];

        // ---------- layer 1: z = Xw1[t] + h1_prev @ U1 ----------
        size_t gmt = (size_t)t * 8 + mt;   // global M-tile of this (t, mt)
        if (t + 1 < TT)
            __builtin_prefetch(Zp + (((size_t)(t + 1) * 8 + mt) * 64 + hb) * 256, 0, 1);

        v8f acc[4];
#pragma unroll
        for (int g = 0; g < 4; ++g)
            acc[g] = *(const v8f*)(Zp + ((gmt * 64 + (g * 16 + hb)) * 32 + lane) * 8);

        if (t > 0) {
            v16h a_cur = *(const v16h*)(h1_prev + laneA);
#pragma unroll
            for (int kt = 0; kt < 8; ++kt) {
                v16h a_nxt = a_cur;
                if (kt < 7) a_nxt = *(const v16h*)(h1_prev + laneA + (size_t)(kt + 1) * 512);
#pragma unroll
                for (int g = 0; g < 4; ++g) {
                    v16h b = *(const v16h*)(smemL + (size_t)(kt * 4 + g) * 512);
                    acc[g] = __builtin_amdgcn_wmma_f32_16x16x32_f16(false, a_cur, false, b,
                                                                    (short)0, acc[g], false, false);
                }
                a_cur = a_nxt;
            }
        }
#pragma unroll
        for (int r = 0; r < 8; ++r) {
            float ig = sigmoidf_(acc[0][r]);
            float fg = sigmoidf_(acc[1][r]);
            float gg = tanhf(acc[2][r]);
            float og = sigmoidf_(acc[3][r]);
            float cn = fmaf(fg, c1[r], ig * gg);
            c1[r] = cn;
            float hh = og * tanhf(cn);
            h1_cur[(((size_t)mt * 8 + kt_h) * 32 + laneS + mo + r) * 16 + eS] = (_Float16)hh;
        }

        // make h1_cur visible across the cluster's WGPs
        __builtin_amdgcn_fence(__ATOMIC_RELEASE, "agent");
        __syncthreads();
        __builtin_amdgcn_s_cluster_barrier();
        __builtin_amdgcn_fence(__ATOMIC_ACQUIRE, "agent");

        // ---------- layer 2: z = h1_cur @ W2 + h2_prev @ U2 + bl2 ----------
#pragma unroll
        for (int g = 0; g < 4; ++g) {
#pragma unroll
            for (int r = 0; r < 8; ++r) acc[g][r] = b2g[g];
        }
        {
            v16h a_cur = *(const v16h*)(h1_cur + laneA);
#pragma unroll
            for (int kt = 0; kt < 8; ++kt) {
                v16h a_nxt = a_cur;
                if (kt < 7)      a_nxt = *(const v16h*)(h1_cur + laneA + (size_t)(kt + 1) * 512);
                else if (t > 0)  a_nxt = *(const v16h*)(h2_prev + laneA);  // preload next gemm
#pragma unroll
                for (int g = 0; g < 4; ++g) {
                    v16h b = *(const v16h*)(smemL + (size_t)(32 + kt * 4 + g) * 512);
                    acc[g] = __builtin_amdgcn_wmma_f32_16x16x32_f16(false, a_cur, false, b,
                                                                    (short)0, acc[g], false, false);
                }
                a_cur = a_nxt;
            }
            if (t > 0) {
#pragma unroll
                for (int kt = 0; kt < 8; ++kt) {
                    v16h a_nxt = a_cur;
                    if (kt < 7) a_nxt = *(const v16h*)(h2_prev + laneA + (size_t)(kt + 1) * 512);
#pragma unroll
                    for (int g = 0; g < 4; ++g) {
                        v16h b = *(const v16h*)(smemL + (size_t)(64 + kt * 4 + g) * 512);
                        acc[g] = __builtin_amdgcn_wmma_f32_16x16x32_f16(false, a_cur, false, b,
                                                                        (short)0, acc[g], false, false);
                    }
                    a_cur = a_nxt;
                }
            }
        }
#pragma unroll
        for (int r = 0; r < 8; ++r) {
            float ig = sigmoidf_(acc[0][r]);
            float fg = sigmoidf_(acc[1][r]);
            float gg = tanhf(acc[2][r]);
            float og = sigmoidf_(acc[3][r]);
            float cn = fmaf(fg, c2[r], ig * gg);
            c2[r] = cn;
            float hh = og * tanhf(cn);
            ysum[r] += hh;
            h2_cur[(((size_t)mt * 8 + kt_h) * 32 + laneS + mo + r) * 16 + eS] = (_Float16)hh;
        }
        // double-buffered h: next iteration's barrier covers remaining deps
    }

#pragma unroll
    for (int r = 0; r < 8; ++r)
        out[(size_t)(m0 + mo + r) * HID + col16] = ysum[r] * (1.0f / (float)TT);
}

// --------------------------------------------------------------------------
extern "C" void kernel_launch(void* const* d_in, const int* in_sizes, int n_in,
                              void* d_out, int out_size, void* d_ws, size_t ws_size,
                              hipStream_t stream) {
    const float* imu = (const float*)d_in[0];
    const float* k1 = (const float*)d_in[1],  *g1 = (const float*)d_in[2],
               * b1 = (const float*)d_in[3],  *m1 = (const float*)d_in[4],
               * v1 = (const float*)d_in[5];
    const float* k2 = (const float*)d_in[6],  *g2 = (const float*)d_in[7],
               * b2 = (const float*)d_in[8],  *m2 = (const float*)d_in[9],
               * v2 = (const float*)d_in[10];
    const float* k3 = (const float*)d_in[11], *g3 = (const float*)d_in[12],
               * b3 = (const float*)d_in[13], *m3 = (const float*)d_in[14],
               * v3 = (const float*)d_in[15];
    const float* W1 = (const float*)d_in[16], *U1 = (const float*)d_in[17],
               * bl1 = (const float*)d_in[18];
    const float* W2 = (const float*)d_in[19], *U2 = (const float*)d_in[20],
               * bl2 = (const float*)d_in[21];

    char* ws = (char*)d_ws;
    size_t off = 0;
    auto alloc = [&](size_t bytes) -> char* {
        char* p = ws + off;
        off += (bytes + 255) & ~(size_t)255;
        return p;
    };

    const size_t NW = (size_t)HID * 4 * HID;  // 262144
    _Float16* W1p = (_Float16*)alloc(NW * 2);
    _Float16* U1p = (_Float16*)alloc(NW * 2);
    _Float16* W2p = (_Float16*)alloc(NW * 2);
    _Float16* U2p = (_Float16*)alloc(NW * 2);
    float* s1 = (float*)alloc(64 * 4);   float* t1 = (float*)alloc(64 * 4);
    float* s2 = (float*)alloc(128 * 4);  float* t2 = (float*)alloc(128 * 4);
    float* s3 = (float*)alloc(256 * 4);  float* t3 = (float*)alloc(256 * 4);
    float*    y1  = (float*)   alloc((size_t)BB * TT * 64 * 4);    // 32 MB
    float*    y2  = (float*)   alloc((size_t)BB * TT * 128 * 4);   // 64 MB
    _Float16* x3p = (_Float16*)alloc((size_t)TT * BB * 256 * 2);   // 64 MB, A-tile packed
    _Float16* h1a = (_Float16*)alloc((size_t)BB * HID * 2);
    _Float16* h1b = (_Float16*)alloc((size_t)BB * HID * 2);
    _Float16* h2a = (_Float16*)alloc((size_t)BB * HID * 2);
    _Float16* h2b = (_Float16*)alloc((size_t)BB * HID * 2);
    float*    Zp  = (float*)   alloc((size_t)TT * BB * 1024 * 4);  // 512 MB, C-tile packed

    k_prep<<<(int)((NW + 255) / 256), 256, 0, stream>>>(
        W1, U1, W2, U2, g1, b1, m1, v1, g2, b2, m2, v2, g3, b3, m3, v3,
        W1p, U1p, W2p, U2p, s1, t1, s2, t2, s3, t3);

    k_conv1<<<(BB * TT * 64) / 256, 256, 0, stream>>>(imu, k1, s1, t1, y1);
    k_conv2<<<(BB * TT * 128) / 256, 256, 0, stream>>>(y1, k2, s2, t2, y2);
    k_conv3<<<(BB * TT * 256) / 256, 256, 0, stream>>>(y2, k3, s3, t3, x3p);

    // 8192 M-tiles * 16 N-groups = 131072 wave-jobs / 8 waves per block
    k_xw1<<<16384, 256, 0, stream>>>(x3p, W1p, bl1, Zp);

    // 16 workgroups (one cluster, block = hidden block) * 8 waves (batch tiles)
    k_lstm<<<16, 256, 0, stream>>>(Zp, U1p, W2p, U2p, bl2,
                                   h1a, h1b, h2a, h2b, (float*)d_out);
}
